// WIRSR_TEFA_Net_41738492182655
// MI455X (gfx1250) — matrix operand order repaired and checked by reading, not compiled, not run on hardware
//
#include <hip/hip_runtime.h>
#include <hip/hip_bf16.h>

#define CHN 192
#define NHD 8
#define HDIM 24
#define ATT_SCALE 0.20412414523193154f  // 24^-0.5
#define CHP 196                          // fp32 LDS row stride

// packed f16 weight workspace layout (offsets in halfs)
#define PK_QKV   0         // [3][8][32][192]  (o padded 24->32 with zeros)
#define PK_WPROJ 147456    // [192][8][32]     (k padded 24->32 with zeros)
#define PK_W1    196608    // [384][192]
#define PK_W2    270336    // [192][384]
#define PK_Q     344064    // [8][32][192]
#define PK_KV    393216    // [2][8][32][192]
#define PK_OPROJ 491520    // [192][8][32]
#define PK_TOTAL 540672

typedef __attribute__((ext_vector_type(16))) _Float16 v16h;
typedef __attribute__((ext_vector_type(8)))  _Float16 v8h;
typedef __attribute__((ext_vector_type(8)))  float    v8f;

__device__ __forceinline__ v8f wmma32(v16h a, v16h b, v8f c) {
  return __builtin_amdgcn_wmma_f32_16x16x32_f16(false, a, false, b, (short)0, c, false, false);
}
__device__ __forceinline__ v16h cat16(v8h a, v8h b) {
  return __builtin_shufflevector(a, b, 0,1,2,3,4,5,6,7,8,9,10,11,12,13,14,15);
}
// A-fragment (16-bit A 16x32): 16 contiguous halves per lane
__device__ __forceinline__ v16h ldA(const _Float16* p, int base, int hi) {
  v8h c0 = *(const v8h*)(p + base + 16 * hi);
  v8h c1 = *(const v8h*)(p + base + 16 * hi + 8);
  return cat16(c0, c1);
}
// B-fragment (operand stored K-contiguous per column): 16 contiguous halves per lane
__device__ __forceinline__ v16h ldB(const _Float16* p, int base, int hi) {
  v8h c0 = *(const v8h*)(p + base + 16 * hi);
  v8h c1 = *(const v8h*)(p + base + 16 * hi + 8);
  return cat16(c0, c1);
}
// B-fragment from raw fp32 weights (fallback path)
__device__ __forceinline__ v16h ldBg(const float* p, int hi) {
  const float* q = p + 16 * hi;
  float4 f0 = *(const float4*)(q);
  float4 f1 = *(const float4*)(q + 4);
  float4 f2 = *(const float4*)(q + 8);
  float4 f3 = *(const float4*)(q + 12);
  v16h r;
  r[0]=(_Float16)f0.x; r[1]=(_Float16)f0.y; r[2]=(_Float16)f0.z; r[3]=(_Float16)f0.w;
  r[4]=(_Float16)f1.x; r[5]=(_Float16)f1.y; r[6]=(_Float16)f1.z; r[7]=(_Float16)f1.w;
  r[8]=(_Float16)f2.x; r[9]=(_Float16)f2.y; r[10]=(_Float16)f2.z; r[11]=(_Float16)f2.w;
  r[12]=(_Float16)f3.x; r[13]=(_Float16)f3.y; r[14]=(_Float16)f3.z; r[15]=(_Float16)f3.w;
  return r;
}
__device__ __forceinline__ v16h ldBg24(const float* p, int hi) {
  const float* q = p + 16 * hi;
  float4 f0 = *(const float4*)(q);
  float4 f1 = *(const float4*)(q + 4);
  v16h r = {};
  r[0]=(_Float16)f0.x; r[1]=(_Float16)f0.y; r[2]=(_Float16)f0.z; r[3]=(_Float16)f0.w;
  r[4]=(_Float16)f1.x; r[5]=(_Float16)f1.y; r[6]=(_Float16)f1.z; r[7]=(_Float16)f1.w;
  if (hi == 0) {
    float4 f2 = *(const float4*)(q + 8);
    float4 f3 = *(const float4*)(q + 12);
    r[8]=(_Float16)f2.x; r[9]=(_Float16)f2.y; r[10]=(_Float16)f2.z; r[11]=(_Float16)f2.w;
    r[12]=(_Float16)f3.x; r[13]=(_Float16)f3.y; r[14]=(_Float16)f3.z; r[15]=(_Float16)f3.w;
  }
  return r;
}
__device__ __forceinline__ int refl(int i) { i = (i < 0) ? -i : i; return (i > 255) ? (510 - i) : i; }

// ======================= weight prepack: fp32 -> padded f16 (once per launch) =======================
__global__ __launch_bounds__(256) void prepack_kernel(
    const float* __restrict__ qkv_w, const float* __restrict__ wproj, const float* __restrict__ w1,
    const float* __restrict__ w2, const float* __restrict__ qw, const float* __restrict__ kvw,
    const float* __restrict__ ow, _Float16* __restrict__ wsh)
{
  int idx = blockIdx.x * 256 + threadIdx.x;
  if (idx >= PK_TOTAL) return;
  _Float16 v;
  if (idx < PK_WPROJ) {                       // qkv [mat][h][o32][k192]
    int j = idx, k = j % 192, r = j / 192, o = r & 31, hh = (r >> 5) & 7, mat = r >> 8;
    v = (o < HDIM) ? (_Float16)qkv_w[(mat * 192 + hh * HDIM + o) * 192 + k] : (_Float16)0.f;
  } else if (idx < PK_W1) {                   // wproj [o192][h8][kb32]
    int j = idx - PK_WPROJ, kb = j & 31, hh = (j >> 5) & 7, o = j >> 8;
    v = (kb < HDIM) ? (_Float16)wproj[o * 192 + hh * HDIM + kb] : (_Float16)0.f;
  } else if (idx < PK_W2) {                   // w1 [384][192]
    v = (_Float16)w1[idx - PK_W1];
  } else if (idx < PK_Q) {                    // w2 [192][384]
    v = (_Float16)w2[idx - PK_W2];
  } else if (idx < PK_KV) {                   // q [h][o32][k192]
    int j = idx - PK_Q, k = j % 192, r = j / 192, o = r & 31, hh = r >> 5;
    v = (o < HDIM) ? (_Float16)qw[(hh * HDIM + o) * 192 + k] : (_Float16)0.f;
  } else if (idx < PK_OPROJ) {                // kv [mat][h][o32][k192]
    int j = idx - PK_KV, k = j % 192, r = j / 192, o = r & 31, hh = (r >> 5) & 7, mat = r >> 8;
    v = (o < HDIM) ? (_Float16)kvw[(mat * 192 + hh * HDIM + o) * 192 + k] : (_Float16)0.f;
  } else {                                    // oproj [o192][h8][kb32]
    int j = idx - PK_OPROJ, kb = j & 31, hh = (j >> 5) & 7, o = j >> 8;
    v = (kb < HDIM) ? (_Float16)ow[o * 192 + hh * HDIM + kb] : (_Float16)0.f;
  }
  wsh[idx] = v;
}

// ======================= Kernel A: window MSA + MLP branch =======================
template <bool P>
__global__ __launch_bounds__(256) void win_kernel(
    const float* __restrict__ x,  const float* __restrict__ g1, const float* __restrict__ b1,
    const float* __restrict__ qkv_w, const float* __restrict__ rpe, const float* __restrict__ wproj_w,
    const float* __restrict__ g2, const float* __restrict__ b2,
    const float* __restrict__ w1, const float* __restrict__ w2,
    const float* __restrict__ alpha_p, const _Float16* __restrict__ wp, float* __restrict__ out)
{
  __shared__ alignas(16) float s_xw[64][CHP];   // fp32 window / residual; front 24KB re-used as f16 LN buffer
  __shared__ float s_mu[64], s_rs[64];
  __shared__ alignas(16) _Float16 s_h[6144];    // 12KB union
  _Float16* hq  = s_h;                 // Q  [64 tok][32 dim]
  _Float16* hk  = s_h + 2048;          // K  [64 tok][32 dim]
  _Float16* hvT = s_h + 4096;          // V^T [32 dim][64 tok]
  _Float16* hp  = s_h;                 // P  [64 q][64 k]   (aliases q,k)
  _Float16* hoh = s_h + 4096;          // out_h [64 tok][32] (aliases vT)
  _Float16* hmid = s_h;                // MLP chunk [64][96]
  _Float16* s_xn = (_Float16*)&s_xw[0][0];      // f16 LN output [64][192] (aliases s_xw)

  const int tid = threadIdx.x, lane = tid & 31, wv = tid >> 5;
  const int hi = lane >> 4, ln16 = lane & 15;
  const int w = blockIdx.x, bb = w >> 10, wy = (w >> 5) & 31, wx = w & 31;
  const int xbase = bb * CHN * 65536;

  for (int idx = tid; idx < 64 * CHN; idx += 256) {
    int c = idx >> 6, m = idx & 63;
    int gy = wy * 8 + (m >> 3), gx = wx * 8 + (m & 7);
    s_xw[m][c] = x[xbase + c * 65536 + gy * 256 + gx];
  }
  __syncthreads();
  if (tid < 64) {
    float s = 0.f, s2 = 0.f;
    for (int c = 0; c < CHN; ++c) { float v = s_xw[tid][c]; s += v; s2 += v * v; }
    float mu = s * (1.f / CHN), var = s2 * (1.f / CHN) - mu * mu;
    s_mu[tid] = mu; s_rs[tid] = rsqrtf(var + 1e-5f);
  }
  __syncthreads();
  // materialize LN1 output as f16 over the same storage (register-staged)
  {
    float tmp[48];
#pragma unroll
    for (int j = 0; j < 48; ++j) {
      int idx = tid + (j << 8);
      int m = idx / CHN, c = idx - m * CHN;
      tmp[j] = (s_xw[m][c] - s_mu[m]) * s_rs[m] * g1[c] + b1[c];
    }
    __syncthreads();
#pragma unroll
    for (int j = 0; j < 48; ++j) s_xn[tid + (j << 8)] = (_Float16)tmp[j];
  }
  __syncthreads();

  v8f zero = {};
  v16h zh = {};
  v8f macc[6];
  for (int j = 0; j < 6; ++j) macc[j] = zero;

  // hoist the wave's QKV A-fragments (head-invariant): tile (qtm,qtn) for all 3 mats / 8 heads
  const int qtm = wv >> 1, qtn = wv & 1;
  const int qol = qtn * 16 + ln16;
  const bool qov = qol < HDIM;
  v16h aA[6];
#pragma unroll
  for (int ks = 0; ks < 6; ++ks) aA[ks] = ldA(s_xn, (qtm * 16 + ln16) * CHN + ks * 32, hi);

  for (int h = 0; h < NHD; ++h) {
    // ---- QKV for this head ----
    for (int mat = 0; mat < 3; ++mat) {
      const float* wr = qkv_w + (mat * CHN + h * HDIM + (qov ? qol : 0)) * CHN;
      v8f acc = zero;
#pragma unroll
      for (int ks = 0; ks < 6; ++ks) {
        v16h bf;
        if (P) bf = ldB(wp, PK_QKV + ((mat * 8 + h) * 32 + qol) * CHN + ks * 32, hi);
        else   bf = qov ? ldBg(wr + ks * 32, hi) : zh;
        acc = wmma32(aA[ks], bf, acc);
      }
      if (mat == 2) {                     // V stored transposed [dim][token]
        v8h hh;
#pragma unroll
        for (int r = 0; r < 8; ++r) hh[r] = (_Float16)acc[r];
        *(v8h*)(hvT + qol * 64 + qtm * 16 + 8 * hi) = hh;
      } else {
        _Float16* dst = (mat == 0) ? hq : hk;
#pragma unroll
        for (int r = 0; r < 8; ++r)
          dst[(qtm * 16 + 8 * hi + r) * 32 + qol] = (_Float16)acc[r];
      }
    }
    __syncthreads();
    // ---- scores transposed: S^T = K x Q^T ----
    v8f sc[2];
    for (int j = 0; j < 2; ++j) {
      int t = wv + 8 * j;
      int tk = t >> 2, tq = t & 3;
      v16h a = ldA(hk, (tk * 16 + ln16) * 32, hi);
      v16h bf = ldB(hq, (tq * 16 + ln16) * 32, hi);
      sc[j] = wmma32(a, bf, zero);
    }
    __syncthreads();                      // q/k reads done before aliased store
    for (int j = 0; j < 2; ++j) {
      int t = wv + 8 * j;
      int tk = t >> 2, tq = t & 3;
      int qy = tq * 16 + ln16;
      int iy = qy >> 3, ix = qy & 7;
      int kb = tk * 16 + 8 * hi;
      v8h hh;
#pragma unroll
      for (int r = 0; r < 8; ++r) {
        int kk = kb + r;
        int ridx = (iy - (kk >> 3) + 7) * 15 + (ix - (kk & 7) + 7);
        hh[r] = (_Float16)(sc[j][r] * ATT_SCALE + rpe[ridx * NHD + h]);
      }
      *(v8h*)(hp + qy * 64 + kb) = hh;
    }
    __syncthreads();
    if (tid < 64) {
      float mx = -1e30f;
      for (int jj = 0; jj < 64; ++jj) { float v = (float)hp[tid * 64 + jj]; mx = v > mx ? v : mx; }
      float sum = 0.f;
      for (int jj = 0; jj < 64; ++jj) { float e = __expf((float)hp[tid * 64 + jj] - mx); sum += e; hp[tid * 64 + jj] = (_Float16)e; }
      float inv = 1.f / sum;
      for (int jj = 0; jj < 64; ++jj) hp[tid * 64 + jj] = (_Float16)((float)hp[tid * 64 + jj] * inv);
    }
    __syncthreads();
    // ---- out_h = P @ V ----
    {
      v8f acc = zero;
      for (int ks = 0; ks < 2; ++ks) {
        v16h a = ldA(hp, (qtm * 16 + ln16) * 64 + ks * 32, hi);
        v16h bf = ldB(hvT, qol * 64 + ks * 32, hi);
        acc = wmma32(a, bf, acc);
      }
      __syncthreads();                    // vT reads done before aliased store
#pragma unroll
      for (int r = 0; r < 8; ++r)
        hoh[(qtm * 16 + 8 * hi + r) * 32 + qol] = (_Float16)acc[r];
    }
    __syncthreads();
    // ---- wproj partial accumulation (K = 24 pad 32) ----
    for (int j = 0; j < 6; ++j) {
      int t = wv + 8 * j, tm = t / 12, tn = t % 12;
      int o = tn * 16 + ln16;
      v16h a = ldA(hoh, (tm * 16 + ln16) * 32, hi);
      v16h bf;
      if (P) bf = ldB(wp, PK_WPROJ + (o * 8 + h) * 32, hi);
      else   bf = ldBg24(wproj_w + o * CHN + h * HDIM, hi);
      macc[j] = wmma32(a, bf, macc[j]);
    }
    __syncthreads();
  }
  // residual: xw1 = xw0 (re-read; f16 alias destroyed fp32 copy) + msa
  for (int j = 0; j < 6; ++j) {
    int t = wv + 8 * j, tm = t / 12, tn = t % 12;
    int c = tn * 16 + ln16;
#pragma unroll
    for (int r = 0; r < 8; ++r) {
      int m = tm * 16 + 8 * hi + r;
      int gy = wy * 8 + (m >> 3), gx = wx * 8 + (m & 7);
      s_xw[m][c] = x[xbase + c * 65536 + gy * 256 + gx] + macc[j][r];
    }
  }
  __syncthreads();
  if (tid < 64) {
    float s = 0.f, s2 = 0.f;
    for (int c = 0; c < CHN; ++c) { float v = s_xw[tid][c]; s += v; s2 += v * v; }
    float mu = s * (1.f / CHN), var = s2 * (1.f / CHN) - mu * mu;
    s_mu[tid] = mu; s_rs[tid] = rsqrtf(var + 1e-5f);
  }
  __syncthreads();
  // preload xw1 into MLP accumulators (xw1 will be overwritten by f16 LN2 buffer)
  v8f m2[6];
  for (int j = 0; j < 6; ++j) {
    int t = wv + 8 * j, tm = t / 12, tn = t % 12;
    int c = tn * 16 + ln16;
#pragma unroll
    for (int r = 0; r < 8; ++r) m2[j][r] = s_xw[tm * 16 + 8 * hi + r][c];
  }
  {
    float tmp[48];
#pragma unroll
    for (int j = 0; j < 48; ++j) {
      int idx = tid + (j << 8);
      int m = idx / CHN, c = idx - m * CHN;
      tmp[j] = (s_xw[m][c] - s_mu[m]) * s_rs[m] * g2[c] + b2[c];
    }
    __syncthreads();
#pragma unroll
    for (int j = 0; j < 48; ++j) s_xn[tid + (j << 8)] = (_Float16)tmp[j];
  }
  __syncthreads();
  // ---- MLP in 4 K-chunks of 96: m2 += gelu(xn2 @ w1^T) @ w2^T ----
  for (int q = 0; q < 4; ++q) {
    v8f racc[3];
    for (int j = 0; j < 3; ++j) {
      int t = wv + 8 * j, tm = t / 6, tn = t % 6;
      int m = tm * 16 + ln16;
      int o = q * 96 + tn * 16 + ln16;
      const float* wr = w1 + o * CHN;
      v8f acc = zero;
      for (int ks = 0; ks < 6; ++ks) {
        v16h a = ldA(s_xn, m * CHN + ks * 32, hi);
        v16h bf;
        if (P) bf = ldB(wp, PK_W1 + o * CHN + ks * 32, hi);
        else   bf = ldBg(wr + ks * 32, hi);
        acc = wmma32(a, bf, acc);
      }
      racc[j] = acc;
    }
    __syncthreads();                      // prev chunk's hmid readers done
    for (int j = 0; j < 3; ++j) {
      int t = wv + 8 * j, tm = t / 6, tn = t % 6;
#pragma unroll
      for (int r = 0; r < 8; ++r) {
        float v = racc[j][r];
        float ge = 0.5f * v * (1.f + erff(v * 0.70710678118f));
        hmid[(tm * 16 + 8 * hi + r) * 96 + tn * 16 + ln16] = (_Float16)ge;
      }
    }
    __syncthreads();
    for (int j = 0; j < 6; ++j) {
      int t = wv + 8 * j, tm = t / 12, tn = t % 12;
      int o = tn * 16 + ln16;
      for (int ks = 0; ks < 3; ++ks) {
        v16h a = ldA(hmid, (tm * 16 + ln16) * 96 + ks * 32, hi);
        v16h bf;
        if (P) bf = ldB(wp, PK_W2 + o * 384 + q * 96 + ks * 32, hi);
        else   bf = ldBg(w2 + o * 384 + q * 96 + ks * 32, hi);
        m2[j] = wmma32(a, bf, m2[j]);
      }
    }
  }
  float alpha = alpha_p[0];
  for (int j = 0; j < 6; ++j) {
    int t = wv + 8 * j, tm = t / 12, tn = t % 12;
    int c = tn * 16 + ln16;
#pragma unroll
    for (int r = 0; r < 8; ++r) {
      int m = tm * 16 + 8 * hi + r;
      int gy = wy * 8 + (m >> 3), gx = wx * 8 + (m & 7);
      int oi = xbase + c * 65536 + gy * 256 + gx;
      out[oi] = x[oi] + alpha * m2[j][r];
    }
  }
}

// ======================= Kernel B: overlapping cross-attention branch =======================
template <bool P>
__global__ __launch_bounds__(256) void oca_kernel(
    const float* __restrict__ x, const float* __restrict__ og, const float* __restrict__ ob,
    const float* __restrict__ qw, const float* __restrict__ kvw, const float* __restrict__ ow,
    const float* __restrict__ alpha_p, const _Float16* __restrict__ wp, float* __restrict__ out)
{
  __shared__ float s_mu[64], s_rs[64];
  __shared__ alignas(16) _Float16 s_q[64 * 32];     // Q  [64 tok][32]
  __shared__ alignas(16) _Float16 s_k[144 * 32];    // K  [144 tok][32]
  __shared__ alignas(16) _Float16 s_vt[32 * 160];   // V^T [32 dim][160 tok] (padded 144->160)
  __shared__ alignas(16) _Float16 s_p[64 * 160];    // P  [64 q][160 k]     (padded 144->160)
  __shared__ alignas(16) _Float16 s_oh[64 * 32];

  const int tid = threadIdx.x, lane = tid & 31, wv = tid >> 5;
  const int hi = lane >> 4, ln16 = lane & 15;
  const int w = blockIdx.x, bb = w >> 10, wy = (w >> 5) & 31, wx = w & 31;
  const int xbase = bb * CHN * 65536;

  if (tid < 64) {
    int gy = wy * 8 + (tid >> 3), gx = wx * 8 + (tid & 7);
    const float* px = x + xbase + gy * 256 + gx;
    float s = 0.f, s2 = 0.f;
    for (int c = 0; c < CHN; ++c) { float v = px[c * 65536]; s += v; s2 += v * v; }
    float mu = s * (1.f / CHN), var = s2 * (1.f / CHN) - mu * mu;
    s_mu[tid] = mu; s_rs[tid] = rsqrtf(var + 1e-5f);
  }
  // zero K-padding columns once (never written afterwards)
  for (int idx = tid; idx < 64 * 16; idx += 256) s_p[(idx >> 4) * 160 + 144 + (idx & 15)] = (_Float16)0.f;
  for (int idx = tid; idx < 32 * 16; idx += 256) s_vt[(idx >> 4) * 160 + 144 + (idx & 15)] = (_Float16)0.f;
  __syncthreads();

  v8f zero = {};
  v16h zh = {};
  v8f macc[6];
  for (int j = 0; j < 6; ++j) macc[j] = zero;

  const int qtm = wv >> 1, qtn = wv & 1;
  const int qol = qtn * 16 + ln16;
  const bool qov = qol < HDIM;

  // hoist head-invariant A operands into registers
  v16h qA[6];
  {
    int m = qtm * 16 + ln16;
    int gy = wy * 8 + (m >> 3), gx = wx * 8 + (m & 7);
    const float* px = x + xbase + gy * 256 + gx;
    float mu = s_mu[m], rs = s_rs[m];
#pragma unroll
    for (int ks = 0; ks < 6; ++ks) {
      v16h a;
#pragma unroll
      for (int i = 0; i < 16; ++i) {
        int ka = ks * 32 + 16 * hi + i;
        a[i] = (_Float16)((px[ka * 65536] - mu) * rs * og[ka] + ob[ka]);
      }
      qA[ks] = a;
    }
  }
  v16h kvA0[6], kvA1[6] = {};
  {
    int mm = wv * 16 + ln16;              // patch row-group g = wv (0..7)
    int rr = mm / 12, cc = mm % 12;
    int py = refl(wy * 8 + rr - 2), pxx = refl(wx * 8 + cc - 2);
    const float* px = x + xbase + py * 256 + pxx;
#pragma unroll
    for (int ks = 0; ks < 6; ++ks) {
      v16h a;
#pragma unroll
      for (int i = 0; i < 16; ++i) {
        int ka = ks * 32 + 16 * hi + i;
        a[i] = (_Float16)px[ka * 65536];
      }
      kvA0[ks] = a;
    }
  }
  if (wv == 0) {                          // wave 0 also owns row-group 8 (tokens 128..143)
    int mm = 128 + ln16;
    int rr = mm / 12, cc = mm % 12;
    int py = refl(wy * 8 + rr - 2), pxx = refl(wx * 8 + cc - 2);
    const float* px = x + xbase + py * 256 + pxx;
#pragma unroll
    for (int ks = 0; ks < 6; ++ks) {
      v16h a;
#pragma unroll
      for (int i = 0; i < 16; ++i) {
        int ka = ks * 32 + 16 * hi + i;
        a[i] = (_Float16)px[ka * 65536];
      }
      kvA1[ks] = a;
    }
  }

  for (int h = 0; h < NHD; ++h) {
    // ---- Q = LN_o(xwin) @ q_w ----
    {
      const float* wr = qw + (h * HDIM + (qov ? qol : 0)) * CHN;
      v8f acc = zero;
#pragma unroll
      for (int ks = 0; ks < 6; ++ks) {
        v16h bf;
        if (P) bf = ldB(wp, PK_Q + (h * 32 + qol) * CHN + ks * 32, hi);
        else   bf = qov ? ldBg(wr + ks * 32, hi) : zh;
        acc = wmma32(qA[ks], bf, acc);
      }
#pragma unroll
      for (int r = 0; r < 8; ++r)
        s_q[(qtm * 16 + 8 * hi + r) * 32 + qol] = (_Float16)acc[r];
    }
    // ---- K,V tiles for this head (reuse hoisted patch fragments) ----
    {
      auto kvgemm = [&](const v16h (&A)[6], int g) {
#pragma unroll
        for (int mt = 0; mt < 4; ++mt) {
          int mat = mt >> 1, tn = mt & 1;
          int ol = tn * 16 + ln16; bool ov = ol < HDIM;
          const float* wr = kvw + (mat * CHN + h * HDIM + (ov ? ol : 0)) * CHN;
          v8f acc = zero;
#pragma unroll
          for (int ks = 0; ks < 6; ++ks) {
            v16h bf;
            if (P) bf = ldB(wp, PK_KV + ((mat * 8 + h) * 32 + ol) * CHN + ks * 32, hi);
            else   bf = ov ? ldBg(wr + ks * 32, hi) : zh;
            acc = wmma32(A[ks], bf, acc);
          }
          if (mat) {
            v8h hh;
#pragma unroll
            for (int r = 0; r < 8; ++r) hh[r] = (_Float16)acc[r];
            *(v8h*)(s_vt + ol * 160 + g * 16 + 8 * hi) = hh;
          } else {
#pragma unroll
            for (int r = 0; r < 8; ++r)
              s_k[(g * 16 + 8 * hi + r) * 32 + ol] = (_Float16)acc[r];
          }
        }
      };
      kvgemm(kvA0, wv);
      if (wv == 0) kvgemm(kvA1, 8);
    }
    __syncthreads();
    // ---- scores transposed: S^T = K x Q^T (144 keys x 64 queries) ----
    for (int j = 0; j < 5; ++j) {
      int t = wv + 8 * j;
      if (t < 36) {
        int tk = t >> 2, tq = t & 3;
        v16h a = ldA(s_k, (tk * 16 + ln16) * 32, hi);
        v16h bf = ldB(s_q, (tq * 16 + ln16) * 32, hi);
        v8f d = wmma32(a, bf, zero);
        int qy = tq * 16 + ln16, kb = tk * 16 + 8 * hi;
        v8h hh;
#pragma unroll
        for (int r = 0; r < 8; ++r) hh[r] = (_Float16)(d[r] * ATT_SCALE);
        *(v8h*)(s_p + qy * 160 + kb) = hh;
      }
    }
    __syncthreads();
    if (tid < 64) {
      float mx = -1e30f;
      for (int jj = 0; jj < 144; ++jj) { float v = (float)s_p[tid * 160 + jj]; mx = v > mx ? v : mx; }
      float sum = 0.f;
      for (int jj = 0; jj < 144; ++jj) { float e = __expf((float)s_p[tid * 160 + jj] - mx); sum += e; s_p[tid * 160 + jj] = (_Float16)e; }
      float inv = 1.f / sum;
      for (int jj = 0; jj < 144; ++jj) s_p[tid * 160 + jj] = (_Float16)((float)s_p[tid * 160 + jj] * inv);
    }
    __syncthreads();
    // ---- out_h = P @ V (K padded to 160, pads zero) ----
    {
      v8f acc = zero;
      for (int ks = 0; ks < 5; ++ks) {
        v16h a = ldA(s_p, (qtm * 16 + ln16) * 160 + ks * 32, hi);
        v16h bf = ldB(s_vt, qol * 160 + ks * 32, hi);
        acc = wmma32(a, bf, acc);
      }
#pragma unroll
      for (int r = 0; r < 8; ++r)
        s_oh[(qtm * 16 + 8 * hi + r) * 32 + qol] = (_Float16)acc[r];
    }
    __syncthreads();
    // ---- oproj partial accumulation ----
    for (int j = 0; j < 6; ++j) {
      int t = wv + 8 * j, tm = t / 12, tn = t % 12;
      int o = tn * 16 + ln16;
      v16h a = ldA(s_oh, (tm * 16 + ln16) * 32, hi);
      v16h bf;
      if (P) bf = ldB(wp, PK_OPROJ + (o * 8 + h) * 32, hi);
      else   bf = ldBg24(ow + o * CHN + h * HDIM, hi);
      macc[j] = wmma32(a, bf, macc[j]);
    }
    __syncthreads();
  }
  float oma = 1.f - alpha_p[0];
  for (int j = 0; j < 6; ++j) {
    int t = wv + 8 * j, tm = t / 12, tn = t % 12;
    int c = tn * 16 + ln16;
#pragma unroll
    for (int r = 0; r < 8; ++r) {
      int m = tm * 16 + 8 * hi + r;
      int gy = wy * 8 + (m >> 3), gx = wx * 8 + (m & 7);
      int oi = xbase + c * 65536 + gy * 256 + gx;
      out[oi] += oma * (macc[j][r] + x[oi]);
    }
  }
}

extern "C" void kernel_launch(void* const* d_in, const int* in_sizes, int n_in,
                              void* d_out, int out_size, void* d_ws, size_t ws_size,
                              hipStream_t stream) {
  const float* x      = (const float*)d_in[0];
  const float* g1     = (const float*)d_in[1];
  const float* b1     = (const float*)d_in[2];
  const float* qkv_w  = (const float*)d_in[3];
  const float* rpe    = (const float*)d_in[4];
  const float* wproj  = (const float*)d_in[5];
  const float* g2     = (const float*)d_in[6];
  const float* b2     = (const float*)d_in[7];
  const float* w1     = (const float*)d_in[8];
  const float* w2     = (const float*)d_in[9];
  const float* alpha  = (const float*)d_in[10];
  const float* og     = (const float*)d_in[11];
  const float* ob     = (const float*)d_in[12];
  const float* q_w    = (const float*)d_in[13];
  const float* kv_w   = (const float*)d_in[14];
  const float* oproj  = (const float*)d_in[15];
  float* out = (float*)d_out;

  const int nwin = 2 * 32 * 32;  // B * (H/8) * (W/8) = 2048
  if (ws_size >= (size_t)PK_TOTAL * sizeof(_Float16)) {
    _Float16* wsh = (_Float16*)d_ws;
    prepack_kernel<<<(PK_TOTAL + 255) / 256, 256, 0, stream>>>(qkv_w, wproj, w1, w2, q_w, kv_w, oproj, wsh);
    win_kernel<true><<<nwin, 256, 0, stream>>>(x, g1, b1, qkv_w, rpe, wproj, g2, b2, w1, w2, alpha, wsh, out);
    oca_kernel<true><<<nwin, 256, 0, stream>>>(x, og, ob, q_w, kv_w, oproj, alpha, wsh, out);
  } else {
    win_kernel<false><<<nwin, 256, 0, stream>>>(x, g1, b1, qkv_w, rpe, wproj, g2, b2, w1, w2, alpha, nullptr, out);
    oca_kernel<false><<<nwin, 256, 0, stream>>>(x, og, ob, q_w, kv_w, oproj, alpha, nullptr, out);
  }
}
